// SimOTA_39161511804929
// MI455X (gfx1250) — matrix operand order repaired and checked by reading, not compiled, not run on hardware
//
#include <hip/hip_runtime.h>
#include <hip/hip_bf16.h>
#include <stdint.h>

#define NUM_CLASSES 80
#define NCAND 10
#define CENTER_RADIUS 2.5f

typedef _Float16 v16h __attribute__((ext_vector_type(16)));
typedef _Float16 v8h  __attribute__((ext_vector_type(8)));
typedef float    v8f  __attribute__((ext_vector_type(8)));
typedef __attribute__((address_space(3))) unsigned char lds_byte;

__device__ __forceinline__ float softplusf(float x) {
    // stable: max(x,0) + log1p(exp(-|x|))
    return fmaxf(x, 0.f) + log1pf(expf(-fabsf(x)));
}

__device__ __forceinline__ uint32_t pkh(float a, float b) {
    unsigned short ua = __builtin_bit_cast(unsigned short, (_Float16)a);
    unsigned short ub = __builtin_bit_cast(unsigned short, (_Float16)b);
    return (uint32_t)ua | ((uint32_t)ub << 16);
}

__device__ __forceinline__ float iou_box(float ax0, float ay0, float ax1, float ay1,
                                         float bx0, float by0, float bx1, float by1) {
    float ix0 = fmaxf(ax0, bx0), iy0 = fmaxf(ay0, by0);
    float ix1 = fminf(ax1, bx1), iy1 = fminf(ay1, by1);
    float iw = fmaxf(ix1 - ix0, 0.f), ih = fmaxf(iy1 - iy0, 0.f);
    float inter = iw * ih;
    float ua = (ax1 - ax0) * (ay1 - ay0) + (bx1 - bx0) * (by1 - by0) - inter;
    return inter / fmaxf(ua, 1e-9f);
}

// ---------------------------------------------------------------------------
// Kernel 1: per 128-anchor tile of one image.
//  Phase 1a: stream 80 logits/anchor once (float4 loads), softplus-sum -> S,
//            pack to f16 pairs into LDS (zero-padded to 96 classes).
//            Also build per-target one-hot f16 rows in LDS (sHot).
//  Phase 1b: z_t = Z(f16) x OneHot(cls) via V_WMMA_F32_16X16X32_F16.
//            A and B fragments are contiguous-in-K per lane half, so each
//            fragment is exactly two ds_load_b128's (no per-element VALU).
//  Phase 2 : per-anchor over all targets: in_box_anchor flag, p_iou,
//            best_t = argmin_t cost (1e9 term constant over t -> dropped).
// ---------------------------------------------------------------------------
__global__ __launch_bounds__(256)
void simota_k1(const float* __restrict__ pred, const float* __restrict__ tgt,
               const float* __restrict__ grid, const float* __restrict__ strd,
               float* __restrict__ gS, int* __restrict__ gInb,
               float* __restrict__ gPiou, int* __restrict__ gBest,
               int* __restrict__ gCnt, int* __restrict__ gMinT,
               int A, int T) {
    __shared__ float sT[64 * 5];                 // target rows (cls,x0,y0,x1,y1)
    __shared__ union {
        _Float16 z16[8][16][96];                 // f16 logits (phase 1 only)
        struct {                                 // phase-2 reduction scratch
            float minc[256]; int argt[256];
            float piou[256]; int flag[256];
        } ph2;
    } u;
    __shared__ _Float16 sHot[64][96];            // one-hot(cls_t) rows, f16
    __shared__ _Float16 sZt[8][16][64];          // WMMA result: z[a, cls_t]
    __shared__ float sS[128];

    const int b   = blockIdx.y;
    const int a0  = blockIdx.x * 128;
    const int tid = threadIdx.x;
    const int w   = tid >> 5;                    // wave id (0..7)
    const int l   = tid & 31;                    // lane id
    const int m   = l & 15;                      // anchor-in-tile (A-matrix M=lane%16)
    const int hh  = l >> 4;                      // class half (0: c<40, 1: c>=40)

    for (int i = tid; i < T * 5; i += 256) sT[i] = tgt[(size_t)b * T * 5 + i];

    // one-hot rows: 64 targets x 96 classes as 48 u32 per row
    {
        uint32_t* hot32 = (uint32_t*)&sHot[0][0];
        for (int i = tid; i < 64 * 48; i += 256) {
            int n = i / 48, kp = i % 48;
            int cls = (n < T) ? (int)tgt[((size_t)b * T + n) * 5] : -1;
            uint32_t v = 0;
            if (2 * kp == cls)     v |= 0x3c00u;
            if (2 * kp + 1 == cls) v |= 0x3c000000u;
            hot32[i] = v;
        }
    }

    // ---- phase 1a: logits -> softplus sum + packed f16 LDS staging -------
    const int aw = a0 + w * 16 + m;
    const bool av = aw < A;
    const float* zrow = pred + ((size_t)b * A + (av ? aw : 0)) * 84 + 4;
    uint32_t* zdst = (uint32_t*)&u.z16[w][m][0];
    float Sp = 0.f;
#pragma unroll
    for (int j4 = 0; j4 < 10; ++j4) {
        int c = hh * 40 + j4 * 4;
        float4 v = av ? *(const float4*)(zrow + c) : make_float4(0.f, 0.f, 0.f, 0.f);
        Sp += softplusf(v.x) + softplusf(v.y) + softplusf(v.z) + softplusf(v.w);
        zdst[c / 2 + 0] = pkh(v.x, v.y);
        zdst[c / 2 + 1] = pkh(v.z, v.w);
    }
    if (hh) {
#pragma unroll
        for (int c = 40; c < 48; ++c) zdst[c] = 0u;   // classes 80..95 = 0
    }
    Sp += __shfl_xor(Sp, 16);
    if (!hh) {
        sS[w * 16 + m] = Sp;
        if (av) gS[(size_t)b * A + aw] = Sp;
    }
    __syncthreads();

    // ---- phase 1b: one-hot GEMM via WMMA ---------------------------------
    // A (16-bit 16x32): lane<16 -> M=lane, K base koffA=0; lane>=16 -> +16.
    //   elements e0..15 = K[base..base+15] (contiguous).
    v16h Afrag[3];
    const int koffA = (l >= 16) ? 16 : 0;
    const _Float16* zr = &u.z16[w][m][0];
#pragma unroll
    for (int ch = 0; ch < 3; ++ch) {
        v8h lo = *(const v8h*)(zr + koffA + ch * 32);
        v8h hi = *(const v8h*)(zr + koffA + ch * 32 + 8);
        Afrag[ch] = __builtin_shufflevector(lo, hi, 0, 1, 2, 3, 4, 5, 6, 7,
                                            8, 9, 10, 11, 12, 13, 14, 15);
    }
    // B (16-bit 32x16): lane<16 -> K=0..15 (+ch*32), lane>=16 -> K=16..31;
    //   all 16 elements contiguous in K.
    const int nloc  = l & 15;
    const int koffB = (l < 16) ? 0 : 16;
#pragma unroll
    for (int n0 = 0; n0 < 4; ++n0) {
        int N = n0 * 16 + nloc;
        const _Float16* hr = &sHot[N][0];
        v8f acc = {0.f, 0.f, 0.f, 0.f, 0.f, 0.f, 0.f, 0.f};
#pragma unroll
        for (int ch = 0; ch < 3; ++ch) {
            v8h blo = *(const v8h*)(hr + koffB + ch * 32);
            v8h bhi = *(const v8h*)(hr + koffB + ch * 32 + 8);
            v16h Bfrag = __builtin_shufflevector(blo, bhi, 0, 1, 2, 3, 4, 5, 6, 7,
                                                 8, 9, 10, 11, 12, 13, 14, 15);
            acc = __builtin_amdgcn_wmma_f32_16x16x32_f16(
                false, Afrag[ch], false, Bfrag, (short)0, acc, false, false);
        }
        // D layout: VGPR r -> M = r (lanes 0-15) / r+8 (lanes 16-31), N = lane%16
#pragma unroll
        for (int r = 0; r < 8; ++r) {
            int mr = (l < 16) ? r : r + 8;
            sZt[w][mr][n0 * 16 + nloc] = (_Float16)acc[r];
        }
    }
    __syncthreads();

    // ---- phase 2: per-anchor, split targets over two thread halves -------
    const int al   = tid & 127;
    const int half = tid >> 7;
    const int aa   = a0 + al;
    const int tsplit = (T + 1) >> 1;
    const int tlo = half ? tsplit : 0;
    const int thi = half ? T : tsplit;

    float minc = 3.4e38f; int argt = tlo; float piou = 0.f; int anyflag = 0;
    if (aa < A) {
        float4 pb = *(const float4*)(pred + ((size_t)b * A + aa) * 84);
        float st = strd[aa];
        float xc = (grid[(size_t)aa * 2 + 0] + 0.5f) * st;
        float yc = (grid[(size_t)aa * 2 + 1] + 0.5f) * st;
        float Sa = sS[al];
        float rad = CENTER_RADIUS * st;
        for (int t = tlo; t < thi; ++t) {
            const float* tr = &sT[t * 5];
            float tx0 = tr[1], ty0 = tr[2], tx1 = tr[3], ty1 = tr[4];
            bool inb = fminf(fminf(xc - tx0, yc - ty0), fminf(tx1 - xc, ty1 - yc)) > 0.f;
            float ctx = (tx0 + tx1) * 0.5f, cty = (ty0 + ty1) * 0.5f;
            bool inc = fmaxf(fabsf(xc - ctx), fabsf(yc - cty)) < rad;
            float iou = iou_box(tx0, ty0, tx1, ty1, pb.x, pb.y, pb.z, pb.w);
            piou = fmaxf(piou, iou);
            anyflag |= (int)(inb || inc);
            float cost = (Sa - (float)sZt[al >> 4][al & 15][t])
                       - 3.f * logf(iou + 1e-8f)
                       + ((inb && inc) ? 0.f : 1.0e5f);   // 1e9 term const over t
            if (cost < minc) { minc = cost; argt = t; }
        }
    }
    u.ph2.minc[tid] = minc; u.ph2.argt[tid] = argt;
    u.ph2.piou[tid] = piou; u.ph2.flag[tid] = anyflag;
    __syncthreads();

    if (tid < 128 && aa < A) {
        float c0 = u.ph2.minc[al], c1 = u.ph2.minc[al + 128];
        int   t0 = u.ph2.argt[al], t1 = u.ph2.argt[al + 128];
        int best = (c1 < c0) ? t1 : t0;               // tie -> lower t (lower half)
        size_t idx = (size_t)b * A + aa;
        gBest[idx] = best;
        gPiou[idx] = fmaxf(u.ph2.piou[al], u.ph2.piou[al + 128]);
        gInb[idx]  = u.ph2.flag[al] | u.ph2.flag[al + 128];
        gCnt[idx]  = 0;
        gMinT[idx] = 0x7fffffff;
    }
}

// ---------------------------------------------------------------------------
// Kernel 2: one block per (image, target). Scans all anchors with pred boxes
// staged through LDS via async global->LDS B128 loads (ASYNCcnt). Maintains
// per-thread sorted top-10 (cost asc / masked-iou desc), merges with 10
// rounds of LDS arg-reductions, then scatters the top dyn_k matches.
// ---------------------------------------------------------------------------
__global__ __launch_bounds__(256)
void simota_k2(const float* __restrict__ pred, const float* __restrict__ tgt,
               const float* __restrict__ grid, const float* __restrict__ strd,
               const float* __restrict__ gS, const int* __restrict__ gInb,
               int* __restrict__ gCnt, int* __restrict__ gMinT,
               int A, int T) {
    __shared__ float sBox[256 * 4];
    __shared__ float redV[256];
    __shared__ int   redI[256];
    __shared__ int   redT[256];
    __shared__ int   selA[NCAND];

    const int blk = blockIdx.x;
    const int b = blk / T;
    const int t = blk % T;
    const int tid = threadIdx.x;

    const float* tr = tgt + ((size_t)b * T + t) * 5;
    const int cls = (int)tr[0];
    const float tx0 = tr[1], ty0 = tr[2], tx1 = tr[3], ty1 = tr[4];
    const float ctx = (tx0 + tx1) * 0.5f, cty = (ty0 + ty1) * 0.5f;

    float cV[NCAND]; int cI[NCAND]; float iV[NCAND];
#pragma unroll
    for (int k = 0; k < NCAND; ++k) { cV[k] = 3.4e38f; cI[k] = -1; iV[k] = -1.f; }

    for (int base = 0; base < A; base += 256) {
        int a = base + tid;
        if (a < A) {
            const float4* gsrc = (const float4*)(pred + ((size_t)b * A + a) * 84);
            lds_byte* ldst = (lds_byte*)(&sBox[tid * 4]);
            // CDNA5 async global->LDS copy (ASYNCcnt-tracked), GV mode.
            asm volatile("global_load_async_to_lds_b128 %0, %1, off"
                         :: "v"(ldst), "v"(gsrc) : "memory");
        }
        asm volatile("s_wait_asynccnt 0" ::: "memory");
        if (a < A) {
            size_t idx = (size_t)b * A + a;
            float px0 = sBox[tid * 4 + 0], py0 = sBox[tid * 4 + 1];
            float px1 = sBox[tid * 4 + 2], py1 = sBox[tid * 4 + 3];
            float iou = iou_box(tx0, ty0, tx1, ty1, px0, py0, px1, py1);
            float st = strd[a];
            float xc = (grid[(size_t)a * 2 + 0] + 0.5f) * st;
            float yc = (grid[(size_t)a * 2 + 1] + 0.5f) * st;
            bool inb = fminf(fminf(xc - tx0, yc - ty0), fminf(tx1 - xc, ty1 - yc)) > 0.f;
            bool inc = fmaxf(fabsf(xc - ctx), fabsf(yc - cty)) < CENTER_RADIUS * st;
            int flag = gInb[idx];
            float z  = pred[idx * 84 + 4 + cls];
            float cost = (gS[idx] - z) - 3.f * logf(iou + 1e-8f)
                       + ((inb && inc) ? 0.f : 1.0e5f)
                       + (flag ? 0.f : 1.0e9f);
            float ioum = flag ? iou : 0.f;
            if (cost < cV[NCAND - 1]) {          // sorted insert (one bubble pass)
                cV[NCAND - 1] = cost; cI[NCAND - 1] = a;
#pragma unroll
                for (int k = NCAND - 1; k > 0; --k) {
                    if (cV[k] < cV[k - 1]) {
                        float tv = cV[k]; cV[k] = cV[k - 1]; cV[k - 1] = tv;
                        int ti = cI[k]; cI[k] = cI[k - 1]; cI[k - 1] = ti;
                    }
                }
            }
            if (ioum > iV[NCAND - 1]) {
                iV[NCAND - 1] = ioum;
#pragma unroll
                for (int k = NCAND - 1; k > 0; --k) {
                    if (iV[k] > iV[k - 1]) {
                        float tv = iV[k]; iV[k] = iV[k - 1]; iV[k - 1] = tv;
                    }
                }
            }
        }
    }
    __syncthreads();

    // ---- global top-10 lowest cost (selection rounds) --------------------
    int pc = 0;
    for (int r = 0; r < NCAND; ++r) {
        redV[tid] = (pc < NCAND) ? cV[pc] : 3.4e38f;
        redI[tid] = (pc < NCAND) ? cI[pc] : -1;
        redT[tid] = tid;
        __syncthreads();
        for (int s = 128; s > 0; s >>= 1) {
            if (tid < s && redV[tid + s] < redV[tid]) {
                redV[tid] = redV[tid + s]; redI[tid] = redI[tid + s]; redT[tid] = redT[tid + s];
            }
            __syncthreads();
        }
        if (tid == redT[0]) pc++;
        if (tid == 0) selA[r] = redI[0];
        __syncthreads();
    }

    // ---- global top-10 masked iou -> dyn_k -------------------------------
    int pi = 0;
    float sumI = 0.f;
    for (int r = 0; r < NCAND; ++r) {
        redV[tid] = (pi < NCAND) ? iV[pi] : -3.4e38f;
        redT[tid] = tid;
        __syncthreads();
        for (int s = 128; s > 0; s >>= 1) {
            if (tid < s && redV[tid + s] > redV[tid]) {
                redV[tid] = redV[tid + s]; redT[tid] = redT[tid + s];
            }
            __syncthreads();
        }
        if (tid == redT[0]) pi++;
        if (tid == 0) sumI += fmaxf(redV[0], 0.f);
        __syncthreads();
    }

    if (tid == 0) {
        int dk = (int)sumI;                       // astype(int32): truncation
        dk = min(max(dk, 1), NCAND);
        for (int k = 0; k < dk; ++k) {
            int a = selA[k];
            if (a >= 0) {
                size_t idx = (size_t)b * A + a;
                if (gInb[idx]) {                  // matching &= in_box_anchor
                    atomicAdd(&gCnt[idx], 1);
                    atomicMin(&gMinT[idx], t);    // argmax(matching) = smallest t
                }
            }
        }
    }
}

// ---------------------------------------------------------------------------
// Kernel 3: finalize per anchor.
//  cnt==0 -> unmatched; cnt==1 -> tp=min_t; cnt>1 (conflict) -> tp=best_t.
// ---------------------------------------------------------------------------
__global__ __launch_bounds__(256)
void simota_k3(const float* __restrict__ tgt, const float* __restrict__ gPiou,
               const int* __restrict__ gBest, const int* __restrict__ gCnt,
               const int* __restrict__ gMinT, float* __restrict__ out,
               int A, int T, int B) {
    size_t BA = (size_t)B * A;
    size_t i = (size_t)blockIdx.x * 256 + threadIdx.x;
    if (i >= BA) return;
    int b = (int)(i / A);
    int cnt = gCnt[i];
    bool mp = cnt > 0;
    int tp = (cnt > 1) ? gBest[i] : ((cnt == 1) ? gMinT[i] : 0);
    tp = min(max(tp, 0), T - 1);
    const float* tr = tgt + ((size_t)b * T + tp) * 5;
    out[i] = mp ? 1.f : 0.f;                                   // mm
    float* bt = out + BA;                                      // box_t
    bt[i * 4 + 0] = mp ? tr[1] : 0.f;
    bt[i * 4 + 1] = mp ? tr[2] : 0.f;
    bt[i * 4 + 2] = mp ? tr[3] : 0.f;
    bt[i * 4 + 3] = mp ? tr[4] : 0.f;
    out[5 * BA + i] = mp ? gPiou[i] : 0.f;                     // obj_t
    out[6 * BA + i] = mp ? tr[0] : (float)NUM_CLASSES;         // cls_t
}

extern "C" void kernel_launch(void* const* d_in, const int* in_sizes, int n_in,
                              void* d_out, int out_size, void* d_ws, size_t ws_size,
                              hipStream_t stream) {
    const float* pred = (const float*)d_in[0];
    const float* tgt  = (const float*)d_in[1];
    const float* grid = (const float*)d_in[2];
    const float* strd = (const float*)d_in[3];
    const int A = in_sizes[2] / 2;
    const int B = (int)((long long)in_sizes[0] / ((long long)A * 84));
    const int T = (int)((long long)in_sizes[1] / ((long long)B * 5));
    const size_t BA = (size_t)B * A;

    char* ws = (char*)d_ws;
    float* gS    = (float*)(ws + BA * 0);
    int*   gInb  = (int*)  (ws + BA * 4);
    float* gPiou = (float*)(ws + BA * 8);
    int*   gBest = (int*)  (ws + BA * 12);
    int*   gCnt  = (int*)  (ws + BA * 16);
    int*   gMinT = (int*)  (ws + BA * 20);
    float* out = (float*)d_out;

    dim3 g1((A + 127) / 128, B);
    simota_k1<<<g1, 256, 0, stream>>>(pred, tgt, grid, strd,
                                      gS, gInb, gPiou, gBest, gCnt, gMinT, A, T);
    simota_k2<<<B * T, 256, 0, stream>>>(pred, tgt, grid, strd,
                                         gS, gInb, gCnt, gMinT, A, T);
    simota_k3<<<(unsigned)((BA + 255) / 256), 256, 0, stream>>>(
        tgt, gPiou, gBest, gCnt, gMinT, out, A, T, B);
}
